// TriAttention_1451698946853
// MI455X (gfx1250) — matrix operand order repaired and checked by reading, not compiled, not run on hardware
//
#include <hip/hip_runtime.h>

// Problem constants (match reference)
#define DIM   1024
#define BATCH 16384
#define KTOK  3
#define NHEAD 16
#define DHEAD 64
#define MQKV  (BATCH * KTOK)   // 49152 rows for q/k/v/out projections

typedef __attribute__((ext_vector_type(16))) __bf16        v16bf;
typedef __attribute__((ext_vector_type(8)))  float         v8f;
typedef __attribute__((ext_vector_type(4)))  unsigned int  v4u;

union Frag { v16bf bf; v4u u4[2]; };

__device__ __forceinline__ unsigned short f2bf(float f) {
  unsigned int u = __float_as_uint(f);
  u += 0x7FFFu + ((u >> 16) & 1u);         // round-to-nearest-even
  return (unsigned short)(u >> 16);
}
__device__ __forceinline__ float bf2f(unsigned short h) {
  return __uint_as_float(((unsigned int)h) << 16);
}

// ---------------------------------------------------------------------------
// fp32 -> bf16 conversion (grid-stride)
// ---------------------------------------------------------------------------
__global__ void cvt_f32_to_bf16(const float* __restrict__ in,
                                unsigned short* __restrict__ out,
                                long long n) {
  long long i = (long long)blockIdx.x * blockDim.x + threadIdx.x;
  long long stride = (long long)gridDim.x * blockDim.x;
  for (; i < n; i += stride) out[i] = f2bf(in[i]);
}

// ---------------------------------------------------------------------------
// Core GEMM tile: one wave computes a 32(M) x 64(N) output tile of
// Y = X(bf16) @ W(bf16)^T + bias using v_wmma_f32_16x16x32_bf16.
// Both A (rows of X) and B (rows of W = columns of W^T) stream contiguously
// along K, using the same contiguous k order for both fragments:
//   A 16x32: lane holds row M=lane&15, K = 16*(lane>=16) .. +15
//   B 32x16: lane holds col N=lane&15, K = 16*(lane>=16) .. +15
//   C/D:     VGPR r, lane L -> row r + 8*(L>=16), col L&15
// ---------------------------------------------------------------------------
template <bool OUT_BF16>
__device__ __forceinline__ void gemm_tile(const unsigned short* __restrict__ X,
                                          const unsigned short* __restrict__ W,
                                          const float* __restrict__ bias,
                                          void* __restrict__ Yv,
                                          int tileM, int tileN) {
  const int lane = threadIdx.x & 31;
  const int lr   = lane & 15;
  const int hi   = lane >> 4;

  const unsigned short* xr0 = X + (size_t)(tileM +      lr) * DIM;
  const unsigned short* xr1 = X + (size_t)(tileM + 16 + lr) * DIM;
  const unsigned short* wr0 = W + (size_t)(tileN +      lr) * DIM;
  const unsigned short* wr1 = W + (size_t)(tileN + 16 + lr) * DIM;
  const unsigned short* wr2 = W + (size_t)(tileN + 32 + lr) * DIM;
  const unsigned short* wr3 = W + (size_t)(tileN + 48 + lr) * DIM;

  const int aoff0 = 16 * hi;           // halves
  const int aoff1 = 16 * hi + 8;
  const int boff  = 16 * hi;

  v8f acc[2][4] = {};

#pragma unroll 8
  for (int k0 = 0; k0 < DIM; k0 += 32) {
    // Prefetch A streams ~256B ahead into all cache levels (WGP scope);
    // weight rows are L2-resident (2MB bf16 per matrix vs 192MB L2).
    __builtin_prefetch(xr0 + k0 + 128, 0, 3);
    __builtin_prefetch(xr1 + k0 + 128, 0, 3);

    Frag a0, a1, b0, b1, b2, b3;
    a0.u4[0] = *(const v4u*)(xr0 + k0 + aoff0);
    a0.u4[1] = *(const v4u*)(xr0 + k0 + aoff1);
    a1.u4[0] = *(const v4u*)(xr1 + k0 + aoff0);
    a1.u4[1] = *(const v4u*)(xr1 + k0 + aoff1);
    b0.u4[0] = *(const v4u*)(wr0 + k0 + boff);
    b0.u4[1] = *(const v4u*)(wr0 + k0 + boff + 8);
    b1.u4[0] = *(const v4u*)(wr1 + k0 + boff);
    b1.u4[1] = *(const v4u*)(wr1 + k0 + boff + 8);
    b2.u4[0] = *(const v4u*)(wr2 + k0 + boff);
    b2.u4[1] = *(const v4u*)(wr2 + k0 + boff + 8);
    b3.u4[0] = *(const v4u*)(wr3 + k0 + boff);
    b3.u4[1] = *(const v4u*)(wr3 + k0 + boff + 8);

    acc[0][0] = __builtin_amdgcn_wmma_f32_16x16x32_bf16(
        false, a0.bf, false, b0.bf, (short)0, acc[0][0], false, false);
    acc[1][0] = __builtin_amdgcn_wmma_f32_16x16x32_bf16(
        false, a1.bf, false, b0.bf, (short)0, acc[1][0], false, false);
    acc[0][1] = __builtin_amdgcn_wmma_f32_16x16x32_bf16(
        false, a0.bf, false, b1.bf, (short)0, acc[0][1], false, false);
    acc[1][1] = __builtin_amdgcn_wmma_f32_16x16x32_bf16(
        false, a1.bf, false, b1.bf, (short)0, acc[1][1], false, false);
    acc[0][2] = __builtin_amdgcn_wmma_f32_16x16x32_bf16(
        false, a0.bf, false, b2.bf, (short)0, acc[0][2], false, false);
    acc[1][2] = __builtin_amdgcn_wmma_f32_16x16x32_bf16(
        false, a1.bf, false, b2.bf, (short)0, acc[1][2], false, false);
    acc[0][3] = __builtin_amdgcn_wmma_f32_16x16x32_bf16(
        false, a0.bf, false, b3.bf, (short)0, acc[0][3], false, false);
    acc[1][3] = __builtin_amdgcn_wmma_f32_16x16x32_bf16(
        false, a1.bf, false, b3.bf, (short)0, acc[1][3], false, false);
  }

#pragma unroll
  for (int j = 0; j < 4; ++j) {
    const int col  = tileN + j * 16 + lr;
    const float bc = bias[col];
#pragma unroll
    for (int i = 0; i < 2; ++i) {
      const int rowb = tileM + i * 16 + 8 * hi;
#pragma unroll
      for (int r = 0; r < 8; ++r) {
        const float val = acc[i][j][r] + bc;
        const size_t idx = (size_t)(rowb + r) * DIM + col;
        if (OUT_BF16) ((unsigned short*)Yv)[idx] = f2bf(val);
        else          ((float*)Yv)[idx] = val;
      }
    }
  }
}

// Generic GEMM: grid (M/64, 1024/256), 256 threads = 8 waves as 2(M) x 4(N).
template <bool OUT_BF16>
__global__ void __launch_bounds__(256)
gemm_bias_kernel(const unsigned short* __restrict__ X,
                 const unsigned short* __restrict__ W,
                 const float* __restrict__ bias,
                 void* __restrict__ Yv) {
  const int wave = threadIdx.x >> 5;
  const int tileM = blockIdx.x * 64 + (wave & 1) * 32;
  const int tileN = blockIdx.y * 256 + (wave >> 1) * 64;
  gemm_tile<OUT_BF16>(X, W, bias, Yv, tileM, tileN);
}

// Fused q/k/v projections. Projection index is blockIdx.x (fastest dim) so
// consecutively-dispatched blocks reuse the same A tile of tokens via L2.
struct QkvArgs {
  const unsigned short* W[3];
  const float*          bias[3];
  unsigned short*       Y[3];
};

__global__ void __launch_bounds__(256)
gemm_qkv_kernel(const unsigned short* __restrict__ X, QkvArgs args) {
  const int proj = blockIdx.x;     // 0..2
  const int wave = threadIdx.x >> 5;
  const int tileM = blockIdx.y * 64 + (wave & 1) * 32;
  const int tileN = blockIdx.z * 256 + (wave >> 1) * 64;
  gemm_tile<true>(X, args.W[proj], args.bias[proj], args.Y[proj], tileM, tileN);
}

// ---------------------------------------------------------------------------
// Attention core: one wave per (b, h). K=3 queries/keys, head_dim 64 ->
// each lane owns 2 head elements. 3x3 scores via wave32 shuffle reduction,
// softmax, weighted sum of context-modulated values. Output bf16.
// ---------------------------------------------------------------------------
__device__ __forceinline__ float wave_sum(float x) {
#pragma unroll
  for (int off = 16; off > 0; off >>= 1) x += __shfl_xor(x, off, 32);
  return x;
}

__global__ void __launch_bounds__(256)
attention_kernel(const unsigned short* __restrict__ qb,
                 const unsigned short* __restrict__ kb,
                 const unsigned short* __restrict__ vb,
                 const unsigned short* __restrict__ cb,
                 unsigned short* __restrict__ ob) {
  const int lane = threadIdx.x & 31;
  const int wid  = blockIdx.x * 8 + (threadIdx.x >> 5);  // 0 .. B*H-1
  const int b = wid >> 4;      // / NHEAD
  const int h = wid & 15;      // % NHEAD
  const size_t colbase = (size_t)h * DHEAD + lane * 2;

  const unsigned int cu = *(const unsigned int*)(cb + (size_t)b * DIM + colbase);
  const float cx = bf2f((unsigned short)(cu & 0xFFFFu));
  const float cy = bf2f((unsigned short)(cu >> 16));

  float qx[3], qy[3], kcx[3], kcy[3], vcx[3], vcy[3];
#pragma unroll
  for (int t = 0; t < 3; ++t) {
    const size_t roff = (size_t)(b * 3 + t) * DIM + colbase;
    const unsigned int qu = *(const unsigned int*)(qb + roff);
    const unsigned int ku = *(const unsigned int*)(kb + roff);
    const unsigned int vu = *(const unsigned int*)(vb + roff);
    qx[t]  = bf2f((unsigned short)(qu & 0xFFFFu));
    qy[t]  = bf2f((unsigned short)(qu >> 16));
    kcx[t] = bf2f((unsigned short)(ku & 0xFFFFu)) * cx;
    kcy[t] = bf2f((unsigned short)(ku >> 16)) * cy;
    vcx[t] = bf2f((unsigned short)(vu & 0xFFFFu)) * cx;
    vcy[t] = bf2f((unsigned short)(vu >> 16)) * cy;
  }

  float s[3][3];
#pragma unroll
  for (int i = 0; i < 3; ++i)
#pragma unroll
    for (int j = 0; j < 3; ++j)
      s[i][j] = wave_sum(qx[i] * kcx[j] + qy[i] * kcy[j]) * 0.125f; // 1/sqrt(64)

#pragma unroll
  for (int i = 0; i < 3; ++i) {
    const float m  = fmaxf(fmaxf(s[i][0], s[i][1]), s[i][2]);
    const float e0 = __expf(s[i][0] - m);
    const float e1 = __expf(s[i][1] - m);
    const float e2 = __expf(s[i][2] - m);
    const float inv = 1.0f / (e0 + e1 + e2);
    const float ox = (e0 * vcx[0] + e1 * vcx[1] + e2 * vcx[2]) * inv;
    const float oy = (e0 * vcy[0] + e1 * vcy[1] + e2 * vcy[2]) * inv;
    const unsigned int packed =
        (unsigned int)f2bf(ox) | ((unsigned int)f2bf(oy) << 16);
    *(unsigned int*)(ob + (size_t)(b * 3 + i) * DIM + colbase) = packed;
  }
}

// ---------------------------------------------------------------------------
// Launcher
// ---------------------------------------------------------------------------
extern "C" void kernel_launch(void* const* d_in, const int* in_sizes, int n_in,
                              void* d_out, int out_size, void* d_ws, size_t ws_size,
                              hipStream_t stream) {
  const float* tokens = (const float*)d_in[0];
  const float* ctx    = (const float*)d_in[1];
  const float* Wq = (const float*)d_in[2];  const float* bq = (const float*)d_in[3];
  const float* Wk = (const float*)d_in[4];  const float* bk = (const float*)d_in[5];
  const float* Wv = (const float*)d_in[6];  const float* bv = (const float*)d_in[7];
  const float* Wc = (const float*)d_in[8];  const float* bc = (const float*)d_in[9];
  const float* Wo = (const float*)d_in[10]; const float* bo = (const float*)d_in[11];

  char* ws = (char*)d_ws;
  size_t off = 0;
  auto carve = [&](size_t bytes) -> void* {
    void* p = ws + off;
    off += (bytes + 255) & ~(size_t)255;
    return p;
  };

  unsigned short* tok_bf  = (unsigned short*)carve((size_t)MQKV * DIM * 2);
  unsigned short* ctx_bf  = (unsigned short*)carve((size_t)BATCH * DIM * 2);
  unsigned short* wq_bf   = (unsigned short*)carve((size_t)DIM * DIM * 2);
  unsigned short* wk_bf   = (unsigned short*)carve((size_t)DIM * DIM * 2);
  unsigned short* wv_bf   = (unsigned short*)carve((size_t)DIM * DIM * 2);
  unsigned short* wc_bf   = (unsigned short*)carve((size_t)DIM * DIM * 2);
  unsigned short* wo_bf   = (unsigned short*)carve((size_t)DIM * DIM * 2);
  unsigned short* q_bf    = (unsigned short*)carve((size_t)MQKV * DIM * 2);
  unsigned short* k_bf    = (unsigned short*)carve((size_t)MQKV * DIM * 2);
  unsigned short* v_bf    = (unsigned short*)carve((size_t)MQKV * DIM * 2);
  unsigned short* c_bf    = (unsigned short*)carve((size_t)BATCH * DIM * 2);
  unsigned short* attn_bf = (unsigned short*)carve((size_t)MQKV * DIM * 2);

  const int CVTB = 4096, CVTT = 256;
  cvt_f32_to_bf16<<<CVTB, CVTT, 0, stream>>>(tokens, tok_bf, (long long)MQKV * DIM);
  cvt_f32_to_bf16<<<CVTB, CVTT, 0, stream>>>(ctx, ctx_bf, (long long)BATCH * DIM);
  cvt_f32_to_bf16<<<256, CVTT, 0, stream>>>(Wq, wq_bf, (long long)DIM * DIM);
  cvt_f32_to_bf16<<<256, CVTT, 0, stream>>>(Wk, wk_bf, (long long)DIM * DIM);
  cvt_f32_to_bf16<<<256, CVTT, 0, stream>>>(Wv, wv_bf, (long long)DIM * DIM);
  cvt_f32_to_bf16<<<256, CVTT, 0, stream>>>(Wc, wc_bf, (long long)DIM * DIM);
  cvt_f32_to_bf16<<<256, CVTT, 0, stream>>>(Wo, wo_bf, (long long)DIM * DIM);

  // Fused q/k/v projections: (49152 x 1024) x (1024 x 1024)^T, 3 weight sets.
  QkvArgs qkv;
  qkv.W[0] = wq_bf; qkv.W[1] = wk_bf; qkv.W[2] = wv_bf;
  qkv.bias[0] = bq; qkv.bias[1] = bk; qkv.bias[2] = bv;
  qkv.Y[0] = q_bf;  qkv.Y[1] = k_bf;  qkv.Y[2] = v_bf;
  dim3 gqkv(3, MQKV / 64, DIM / 256);
  gemm_qkv_kernel<<<gqkv, 256, 0, stream>>>(tok_bf, qkv);

  // ctx projection: (16384 x 1024)
  dim3 gc(BATCH / 64, DIM / 256);
  gemm_bias_kernel<true><<<gc, 256, 0, stream>>>(ctx_bf, wc_bf, bc, c_bf);

  // attention core: one wave per (b, h)
  attention_kernel<<<(BATCH * NHEAD) / 8, 256, 0, stream>>>(q_bf, k_bf, v_bf,
                                                            c_bf, attn_bf);

  // output projection -> f32 d_out
  dim3 go(MQKV / 64, DIM / 256);
  gemm_bias_kernel<false><<<go, 256, 0, stream>>>(attn_bf, wo_bf, bo,
                                                  (float*)d_out);
}
